// VanillaMPN_46256797778077
// MI455X (gfx1250) — matrix operand order repaired and checked by reading, not compiled, not run on hardware
//
#include <hip/hip_runtime.h>

// ---------------------------------------------------------------------------
// VanillaMPN forward for gfx1250 (MI455X), bf16 WMMA (f32 accumulate).
// All GEMMs run through V_WMMA_F32_16X16X32_BF16. Concats are virtualized in
// the A-operand gather; segment_sum is fused as f32 global atomics.
// Round 2: MT row-tiles per wave to reuse B fragments (vmem/wmma from
// 2/NT+2 down to 2/NT+2/MT) and amortize s_wait_loadcnt over MT WMMAs.
// ---------------------------------------------------------------------------

typedef __attribute__((ext_vector_type(16))) __bf16 v16bf;
typedef __attribute__((ext_vector_type(8)))  __bf16 v8bf;
typedef __attribute__((ext_vector_type(8)))  float  v8f;

// A-fragment (16x32 bf16, row-major source). Lane L holds row M=(L&15);
// lanes<16 cover K {kb+0..7, kb+16..23}, lanes>=16 cover K {kb+8..15, kb+24..31}.
static __device__ __forceinline__ v16bf load_a_frag(const __bf16* arow, int kb, int s) {
  v8bf lo = *(const v8bf*)(arow + kb + s);
  v8bf hi = *(const v8bf*)(arow + kb + 16 + s);
  return __builtin_shufflevector(lo, hi, 0,1,2,3,4,5,6,7,8,9,10,11,12,13,14,15);
}

// Pack fp32 weights [Kr,Nr] row-major into B-operand tiles:
// dst[((kt*(Np/16)+nt)*32 + lane)*16 + e] = W[kt*32 + 8*(lane>=16) + e + 8*(e>=8), nt*16 + (lane&15)]
// Zero-pads K->Kp, N->Np.
__global__ __launch_bounds__(256) void pack_w_kernel(const float* __restrict__ W,
                                                     __bf16* __restrict__ dst,
                                                     int Kr, int Nr, int Kp, int Np) {
  int tid = blockIdx.x * blockDim.x + threadIdx.x;
  int total = Kp * Np;
  if (tid >= total) return;
  int e    = tid & 15;
  int lane = (tid >> 4) & 31;
  int t    = tid >> 9;             // tile index = kt*(Np/16) + nt
  int NT   = Np >> 4;
  int nt   = t % NT;
  int kt   = t / NT;
  int k = kt * 32 + ((lane >= 16) ? 8 : 0) + e + ((e >= 8) ? 8 : 0);
  int n = nt * 16 + (lane & 15);
  float v = (k < Kr && n < Nr) ? W[(long)k * Nr + n] : 0.0f;
  dst[tid] = (__bf16)v;
}

// fp32 [M,Cin] -> bf16 [M,Cpad] with zero pad.
__global__ __launch_bounds__(256) void cvt_pad_kernel(const float* __restrict__ src,
                                                      __bf16* __restrict__ dst,
                                                      long M, int Cin, int Cpad) {
  long tid = (long)blockIdx.x * blockDim.x + threadIdx.x;
  if (tid >= M * (long)Cpad) return;
  long r = tid / Cpad;
  int  c = (int)(tid % Cpad);
  dst[tid] = (c < Cin) ? (__bf16)src[r * (long)Cin + c] : (__bf16)0.0f;
}

__global__ __launch_bounds__(256) void f32_to_bf16_kernel(const float* __restrict__ s,
                                                          __bf16* __restrict__ d, long n) {
  long i = (long)blockIdx.x * blockDim.x + threadIdx.x;
  if (i < n) d[i] = (__bf16)s[i];
}

__global__ __launch_bounds__(256) void zero_f32_kernel(float* __restrict__ p, long n) {
  long i = (long)blockIdx.x * blockDim.x + threadIdx.x;
  if (i < n) p[i] = 0.0f;
}

// Generic GEMM: out = act(A[M,K](bf16) @ Wp(packed bf16,[K,Np]) + bias).
// One wave per MT consecutive 16-row tiles; NT = Np/16 accumulator tiles per
// row tile. Each B fragment is reused by MT WMMAs.
template<int NT, int MT, bool RELU, bool OBF, bool OF>
__global__ __launch_bounds__(256) void gemm_kernel(
    const __bf16* __restrict__ A, const __bf16* __restrict__ Wp,
    const float* __restrict__ bias,
    __bf16* __restrict__ Ob, float* __restrict__ Of,
    long M, int K, int ldo, int Nreal)
{
  const int  lane = threadIdx.x & 31;
  const int  wave = threadIdx.x >> 5;
  const long m0   = ((long)blockIdx.x * 8 + wave) * (16 * MT);
  if (m0 >= M) return;                       // wave-uniform: EXEC stays full
  const int half = lane >> 4;
  const int s    = half * 8;
  const __bf16* arow[MT];
  #pragma unroll
  for (int mt = 0; mt < MT; ++mt) {
    long row = m0 + mt * 16 + (lane & 15);
    if (row >= M) row = M - 1;               // clamp loads; stores are guarded
    arow[mt] = A + row * (long)K;
  }
  v8f z = {0,0,0,0,0,0,0,0};
  v8f acc[MT][NT];
  #pragma unroll
  for (int mt = 0; mt < MT; ++mt)
    #pragma unroll
    for (int t = 0; t < NT; ++t) acc[mt][t] = z;

  for (int kb = 0; kb < K; kb += 32) {
    v16bf a[MT];
    #pragma unroll
    for (int mt = 0; mt < MT; ++mt) a[mt] = load_a_frag(arow[mt], kb, s);
    const __bf16* wb = Wp + ((long)(kb >> 5) * NT) * 512 + lane * 16;
    #pragma unroll
    for (int t = 0; t < NT; ++t) {
      v16bf b = *(const v16bf*)(wb + t * 512);
      #pragma unroll
      for (int mt = 0; mt < MT; ++mt)
        acc[mt][t] = __builtin_amdgcn_wmma_f32_16x16x32_bf16(
            false, a[mt], false, b, (short)0, acc[mt][t], false, false);
    }
  }
  #pragma unroll
  for (int mt = 0; mt < MT; ++mt) {
    #pragma unroll
    for (int t = 0; t < NT; ++t) {
      const int col = t * 16 + (lane & 15);
      if (col >= Nreal) continue;
      const float bv = bias ? bias[col] : 0.0f;
      #pragma unroll
      for (int r = 0; r < 8; ++r) {
        long rr = m0 + mt * 16 + r + 8 * half;   // C layout: M = r + 8*(lane>>4)
        if (rr < M) {
          float v = acc[mt][t][r] + bv;
          if (RELU) v = fmaxf(v, 0.0f);
          if (OBF) Ob[rr * (long)ldo + col] = (__bf16)v;
          if (OF)  Of[rr * (long)ldo + col] = v;
        }
      }
    }
  }
}

// mlp_edge layer 1: A row e = [nf0[dst] | nf[dst] | nf0[src] | nf[src] | ef0[e] | ef[e]],
// K = 640, N = 64 (NT=4), MT=2 row tiles per wave. All concat boundaries are
// multiples of the K=32 WMMA step, so each K-chunk reads exactly one source.
__global__ __launch_bounds__(256) void edge_mlp_fused_kernel(
    const __bf16* __restrict__ nf0, const __bf16* __restrict__ nf,
    const __bf16* __restrict__ ef0, const __bf16* __restrict__ ef,
    const int* __restrict__ srcI, const int* __restrict__ dstI,
    const __bf16* __restrict__ Wp, const float* __restrict__ bias,
    __bf16* __restrict__ out, long E)
{
  const int  lane = threadIdx.x & 31;
  const int  wave = threadIdx.x >> 5;
  const long m0   = ((long)blockIdx.x * 8 + wave) * 32;   // MT=2
  if (m0 >= E) return;
  const int half = lane >> 4;
  const int s    = half * 8;
  const __bf16* segs[2][6];
  #pragma unroll
  for (int mt = 0; mt < 2; ++mt) {
    long row = m0 + mt * 16 + (lane & 15);
    if (row >= E) row = E - 1;
    const long d  = dstI[row];
    const long sj = srcI[row];
    segs[mt][0] = nf0 + d  * 128;
    segs[mt][1] = nf  + d  * 128;
    segs[mt][2] = nf0 + sj * 128;
    segs[mt][3] = nf  + sj * 128;
    segs[mt][4] = ef0 + row * 64;
    segs[mt][5] = ef  + row * 64;
  }
  v8f z = {0,0,0,0,0,0,0,0};
  v8f acc[2][4];
  #pragma unroll
  for (int mt = 0; mt < 2; ++mt)
    #pragma unroll
    for (int t = 0; t < 4; ++t) acc[mt][t] = z;

  #pragma unroll
  for (int kb = 0; kb < 640; kb += 32) {
    int si, o;
    if      (kb < 128) { si = 0; o = kb;       }
    else if (kb < 256) { si = 1; o = kb - 128; }
    else if (kb < 384) { si = 2; o = kb - 256; }
    else if (kb < 512) { si = 3; o = kb - 384; }
    else if (kb < 576) { si = 4; o = kb - 512; }
    else               { si = 5; o = kb - 576; }
    v16bf a0 = load_a_frag(segs[0][si], o, s);
    v16bf a1 = load_a_frag(segs[1][si], o, s);
    const __bf16* wb = Wp + ((long)(kb >> 5) * 4) * 512 + lane * 16;
    #pragma unroll
    for (int t = 0; t < 4; ++t) {
      v16bf b = *(const v16bf*)(wb + t * 512);
      acc[0][t] = __builtin_amdgcn_wmma_f32_16x16x32_bf16(
          false, a0, false, b, (short)0, acc[0][t], false, false);
      acc[1][t] = __builtin_amdgcn_wmma_f32_16x16x32_bf16(
          false, a1, false, b, (short)0, acc[1][t], false, false);
    }
  }
  #pragma unroll
  for (int mt = 0; mt < 2; ++mt) {
    #pragma unroll
    for (int t = 0; t < 4; ++t) {
      const int col  = t * 16 + (lane & 15);
      const float bv = bias[col];
      #pragma unroll
      for (int r = 0; r < 8; ++r) {
        long rr = m0 + mt * 16 + r + 8 * half;
        if (rr < E) {
          float v = fmaxf(acc[mt][t][r] + bv, 0.0f);
          out[rr * 64 + col] = (__bf16)v;
        }
      }
    }
  }
}

// mlp_node + segment_sum: A row e = [nf0[dst] | nf[dst] | ef_new[e]], K = 320,
// N = 128 (NT=8), MT=2. ReLU(out) atomically accumulated into nfacc[dst[e],:].
__global__ __launch_bounds__(256) void node_msg_kernel(
    const __bf16* __restrict__ nf0, const __bf16* __restrict__ nf,
    const __bf16* __restrict__ efn, const int* __restrict__ dstI,
    const __bf16* __restrict__ Wp, const float* __restrict__ bias,
    float* __restrict__ nfacc, long E)
{
  const int  lane = threadIdx.x & 31;
  const int  wave = threadIdx.x >> 5;
  const long m0   = ((long)blockIdx.x * 8 + wave) * 32;   // MT=2
  if (m0 >= E) return;
  const int half = lane >> 4;
  const int s    = half * 8;
  const __bf16* segs[2][3];
  #pragma unroll
  for (int mt = 0; mt < 2; ++mt) {
    long row = m0 + mt * 16 + (lane & 15);
    if (row >= E) row = E - 1;
    const long d = dstI[row];
    segs[mt][0] = nf0 + d * 128;
    segs[mt][1] = nf  + d * 128;
    segs[mt][2] = efn + row * 64;
  }
  v8f z = {0,0,0,0,0,0,0,0};
  v8f acc[2][8];
  #pragma unroll
  for (int mt = 0; mt < 2; ++mt)
    #pragma unroll
    for (int t = 0; t < 8; ++t) acc[mt][t] = z;

  #pragma unroll
  for (int kb = 0; kb < 320; kb += 32) {
    int si, o;
    if      (kb < 128) { si = 0; o = kb;       }
    else if (kb < 256) { si = 1; o = kb - 128; }
    else               { si = 2; o = kb - 256; }
    v16bf a0 = load_a_frag(segs[0][si], o, s);
    v16bf a1 = load_a_frag(segs[1][si], o, s);
    const __bf16* wb = Wp + ((long)(kb >> 5) * 8) * 512 + lane * 16;
    #pragma unroll
    for (int t = 0; t < 8; ++t) {
      v16bf b = *(const v16bf*)(wb + t * 512);
      acc[0][t] = __builtin_amdgcn_wmma_f32_16x16x32_bf16(
          false, a0, false, b, (short)0, acc[0][t], false, false);
      acc[1][t] = __builtin_amdgcn_wmma_f32_16x16x32_bf16(
          false, a1, false, b, (short)0, acc[1][t], false, false);
    }
  }
  int nd[2][8];
  #pragma unroll
  for (int mt = 0; mt < 2; ++mt)
    #pragma unroll
    for (int r = 0; r < 8; ++r) {
      long rr = m0 + mt * 16 + r + 8 * half;
      nd[mt][r] = (rr < E) ? dstI[rr] : -1;
    }
  #pragma unroll
  for (int mt = 0; mt < 2; ++mt) {
    #pragma unroll
    for (int t = 0; t < 8; ++t) {
      const int col  = t * 16 + (lane & 15);
      const float bv = bias[col];
      #pragma unroll
      for (int r = 0; r < 8; ++r) {
        if (nd[mt][r] >= 0) {
          float v = fmaxf(acc[mt][t][r] + bv, 0.0f);
          atomicAdd(nfacc + (long)nd[mt][r] * 128 + col, v);
        }
      }
    }
  }
}

// ---------------------------------------------------------------------------
// Host orchestration
// ---------------------------------------------------------------------------
extern "C" void kernel_launch(void* const* d_in, const int* in_sizes, int n_in,
                              void* d_out, int out_size, void* d_ws, size_t ws_size,
                              hipStream_t stream) {
  (void)n_in; (void)out_size; (void)ws_size;
  const long Nn = in_sizes[0] / 256;   // nodes
  const long Ee = in_sizes[1] / 17;    // edges

  const float* x    = (const float*)d_in[0];
  const float* ea   = (const float*)d_in[1];
  const int*   eidx = (const int*)d_in[2];
  const int*   srcI = eidx;            // edge_index[0] (j / source)
  const int*   dstI = eidx + Ee;       // edge_index[1] (i / target)

  // params in setup_inputs() insertion order: (W,b) pairs
  const float* neW0 = (const float*)d_in[3];  const float* neB0 = (const float*)d_in[4];
  const float* neW1 = (const float*)d_in[5];  const float* neB1 = (const float*)d_in[6];
  const float* neW2 = (const float*)d_in[7];  const float* neB2 = (const float*)d_in[8];
  const float* eeW0 = (const float*)d_in[9];  const float* eeB0 = (const float*)d_in[10];
  const float* eeW1 = (const float*)d_in[11]; const float* eeB1 = (const float*)d_in[12];
  const float* eeW2 = (const float*)d_in[13]; const float* eeB2 = (const float*)d_in[14];
  const float* eeW3 = (const float*)d_in[15]; const float* eeB3 = (const float*)d_in[16];
  const float* meW0 = (const float*)d_in[17]; const float* meB0 = (const float*)d_in[18];
  const float* meW1 = (const float*)d_in[19]; const float* meB1 = (const float*)d_in[20];
  const float* mnW0 = (const float*)d_in[21]; const float* mnB0 = (const float*)d_in[22];
  const float* cW0  = (const float*)d_in[23]; const float* cB0  = (const float*)d_in[24];
  const float* cW1  = (const float*)d_in[25]; const float* cB1  = (const float*)d_in[26];
  const float* cW2  = (const float*)d_in[27]; const float* cB2  = (const float*)d_in[28];

  // workspace bump allocator (256B aligned regions)
  char* wsp = (char*)d_ws;
  auto alloc = [&](size_t bytes) -> char* {
    char* r = wsp;
    wsp += (bytes + 255) & ~(size_t)255;
    return r;
  };
  // packed weights (bf16)
  __bf16* wp_ne0 = (__bf16*)alloc(32768 * 2);   // 256x128
  __bf16* wp_ne1 = (__bf16*)alloc(8192 * 2);    // 128x64
  __bf16* wp_ne2 = (__bf16*)alloc(8192 * 2);    // 64x128
  __bf16* wp_ee0 = (__bf16*)alloc(1024 * 2);    // 32(pad17)x32
  __bf16* wp_ee1 = (__bf16*)alloc(2048 * 2);    // 32x64
  __bf16* wp_ee2 = (__bf16*)alloc(4096 * 2);    // 64x64
  __bf16* wp_ee3 = (__bf16*)alloc(4096 * 2);    // 64x64
  __bf16* wp_me0 = (__bf16*)alloc(40960 * 2);   // 640x64
  __bf16* wp_me1 = (__bf16*)alloc(4096 * 2);    // 64x64
  __bf16* wp_mn  = (__bf16*)alloc(40960 * 2);   // 320x128
  __bf16* wp_c0  = (__bf16*)alloc(4096 * 2);    // 64x64
  __bf16* wp_c1  = (__bf16*)alloc(2048 * 2);    // 64x32
  __bf16* wp_c2  = (__bf16*)alloc(512 * 2);     // 32x16(pad N 1->16)
  // activations
  __bf16* x16   = (__bf16*)alloc((size_t)Nn * 256 * 2);
  __bf16* h1    = (__bf16*)alloc((size_t)Nn * 128 * 2);
  __bf16* h2    = (__bf16*)alloc((size_t)Nn * 64 * 2);
  __bf16* nfA   = (__bf16*)alloc((size_t)Nn * 128 * 2);
  __bf16* nfB   = (__bf16*)alloc((size_t)Nn * 128 * 2);
  __bf16* nf0   = (__bf16*)alloc((size_t)Nn * 128 * 2);
  float*  nfacc = (float*) alloc((size_t)Nn * 128 * 4);
  __bf16* ea16  = (__bf16*)alloc((size_t)Ee * 32 * 2);
  __bf16* efA   = (__bf16*)alloc((size_t)Ee * 64 * 2);
  __bf16* efB   = (__bf16*)alloc((size_t)Ee * 64 * 2);
  __bf16* ef0   = (__bf16*)alloc((size_t)Ee * 64 * 2);
  __bf16* t64a  = (__bf16*)alloc((size_t)Ee * 64 * 2);
  __bf16* t64b  = (__bf16*)alloc((size_t)Ee * 64 * 2);

  auto packW = [&](const float* W, __bf16* dst, int Kr, int Nr, int Kp, int Np) {
    int total = Kp * Np;
    pack_w_kernel<<<(total + 255) / 256, 256, 0, stream>>>(W, dst, Kr, Nr, Kp, Np);
  };
  packW(neW0, wp_ne0, 256, 128, 256, 128);
  packW(neW1, wp_ne1, 128, 64, 128, 64);
  packW(neW2, wp_ne2, 64, 128, 64, 128);
  packW(eeW0, wp_ee0, 17, 32, 32, 32);
  packW(eeW1, wp_ee1, 32, 64, 32, 64);
  packW(eeW2, wp_ee2, 64, 64, 64, 64);
  packW(eeW3, wp_ee3, 64, 64, 64, 64);
  packW(meW0, wp_me0, 640, 64, 640, 64);
  packW(meW1, wp_me1, 64, 64, 64, 64);
  packW(mnW0, wp_mn, 320, 128, 320, 128);
  packW(cW0, wp_c0, 64, 64, 64, 64);
  packW(cW1, wp_c1, 64, 32, 64, 32);
  packW(cW2, wp_c2, 32, 1, 32, 16);

  // input conversions (K of edge_attr padded 17 -> 32)
  {
    long tot = Nn * 256;
    cvt_pad_kernel<<<(int)((tot + 255) / 256), 256, 0, stream>>>(x, x16, Nn, 256, 256);
    tot = Ee * 32;
    cvt_pad_kernel<<<(int)((tot + 255) / 256), 256, 0, stream>>>(ea, ea16, Ee, 17, 32);
  }

  // grid sizing: waves = ceil(M / (16*MT)), 8 waves per block
  auto gblk = [](long M, int MT) {
    long waves = (M + 16L * MT - 1) / (16L * MT);
    return (int)((waves + 7) / 8);
  };
  const int nb2 = gblk(Nn, 2), nb4 = gblk(Nn, 4);
  const int eb2 = gblk(Ee, 2), eb4 = gblk(Ee, 4);

  // node embedding: 256 ->128(relu) ->64(relu) ->128
  gemm_kernel<8, 2, true,  true, false><<<nb2, 256, 0, stream>>>(x16, wp_ne0, neB0, h1,  nullptr, Nn, 256, 128, 128);
  gemm_kernel<4, 4, true,  true, false><<<nb4, 256, 0, stream>>>(h1,  wp_ne1, neB1, h2,  nullptr, Nn, 128, 64,  64);
  gemm_kernel<8, 2, false, true, false><<<nb2, 256, 0, stream>>>(h2,  wp_ne2, neB2, nfA, nullptr, Nn, 64,  128, 128);
  hipMemcpyAsync(nf0, nfA, (size_t)Nn * 128 * 2, hipMemcpyDeviceToDevice, stream);

  // edge embedding: 17(pad32) ->32(relu) ->64(relu) ->64(relu) ->64
  gemm_kernel<2, 4, true,  true, false><<<eb4, 256, 0, stream>>>(ea16, wp_ee0, eeB0, t64b, nullptr, Ee, 32, 32, 32);
  gemm_kernel<4, 4, true,  true, false><<<eb4, 256, 0, stream>>>(t64b, wp_ee1, eeB1, t64a, nullptr, Ee, 32, 64, 64);
  gemm_kernel<4, 4, true,  true, false><<<eb4, 256, 0, stream>>>(t64a, wp_ee2, eeB2, t64b, nullptr, Ee, 64, 64, 64);
  gemm_kernel<4, 4, false, true, false><<<eb4, 256, 0, stream>>>(t64b, wp_ee3, eeB3, efA,  nullptr, Ee, 64, 64, 64);
  hipMemcpyAsync(ef0, efA, (size_t)Ee * 64 * 2, hipMemcpyDeviceToDevice, stream);

  __bf16* nf_cur = nfA; __bf16* nf_alt = nfB;
  __bf16* ef_cur = efA; __bf16* ef_alt = efB;
  for (int step = 0; step < 4; ++step) {
    // ef_new = relu(relu([x_i|x_j|ef_c] W0 + b0) W1 + b1)
    edge_mlp_fused_kernel<<<eb2, 256, 0, stream>>>(nf0, nf_cur, ef0, ef_cur, srcI, dstI,
                                                   wp_me0, meB0, t64a, Ee);
    gemm_kernel<4, 4, true, true, false><<<eb4, 256, 0, stream>>>(t64a, wp_me1, meB1, ef_alt,
                                                                  nullptr, Ee, 64, 64, 64);
    // nf_new = segment_sum(relu([x_i|ef_new] Wn + bn), dst)
    long accn = Nn * 128;
    zero_f32_kernel<<<(int)((accn + 255) / 256), 256, 0, stream>>>(nfacc, accn);
    node_msg_kernel<<<eb2, 256, 0, stream>>>(nf0, nf_cur, ef_alt, dstI, wp_mn, mnB0, nfacc, Ee);
    f32_to_bf16_kernel<<<(int)((accn + 255) / 256), 256, 0, stream>>>(nfacc, nf_alt, accn);
    { __bf16* t = nf_cur; nf_cur = nf_alt; nf_alt = t; }
    { __bf16* t = ef_cur; ef_cur = ef_alt; ef_alt = t; }
  }

  // classifier: 64 ->64(relu) ->32(relu) ->1, f32 output
  gemm_kernel<4, 4, true,  true,  false><<<eb4, 256, 0, stream>>>(ef_cur, wp_c0, cB0, t64a, nullptr, Ee, 64, 64, 64);
  gemm_kernel<2, 4, true,  true,  false><<<eb4, 256, 0, stream>>>(t64a,   wp_c1, cB1, t64b, nullptr, Ee, 64, 32, 32);
  gemm_kernel<1, 4, false, false, true ><<<eb4, 256, 0, stream>>>(t64b,   wp_c2, cB2, nullptr,
                                                                  (float*)d_out, Ee, 32, 1, 1);
}
